// RNNT_11931419148511
// MI455X (gfx1250) — matrix operand order repaired and checked by reading, not compiled, not run on hardware
//
#include <hip/hip_runtime.h>
#include <hip/hip_bf16.h>
#include <math.h>

typedef __attribute__((ext_vector_type(16))) _Float16 v16h;
typedef __attribute__((ext_vector_type(8)))  _Float16 v8h;
typedef __attribute__((ext_vector_type(8)))  float    v8f;

#define B_   8
#define T_   256
#define L_   63
#define U_   64
#define V_   512
#define DENC 512
#define DDEC 256
#define DJ   256

// ---------------------------------------------------------------- embedding
__global__ void k_embed(const float* __restrict__ embed, const int* __restrict__ ys,
                        float* __restrict__ x0) {
    int idx = blockIdx.x * 256 + threadIdx.x;      // B*U*DDEC total
    int k  = idx & (DDEC - 1);
    int bu = idx >> 8;                              // DDEC == 256
    int u  = bu & (U_ - 1);
    int b  = bu >> 6;                               // U_ == 64
    int tok = (u == 0) ? 0 : ys[b * L_ + (u - 1)];
    x0[idx] = embed[tok * DDEC + k];
}

// ---------------------------------------------------------------- LSTM layer
__global__ void __launch_bounds__(DDEC) k_lstm(const float* __restrict__ x,
        const float* __restrict__ Wih, const float* __restrict__ Whh,
        const float* __restrict__ bih, const float* __restrict__ bhh,
        float* __restrict__ out) {
    __shared__ float xt[DDEC];
    __shared__ float hs[DDEC];
    int b = blockIdx.x;
    int j = threadIdx.x;
    float c = 0.f, hprev = 0.f;
    float b0 = bih[j]            + bhh[j];
    float b1 = bih[DDEC + j]     + bhh[DDEC + j];
    float b2 = bih[2 * DDEC + j] + bhh[2 * DDEC + j];
    float b3 = bih[3 * DDEC + j] + bhh[3 * DDEC + j];
    const float* wi0 = Wih + (size_t)(0 * DDEC + j) * DDEC;
    const float* wi1 = Wih + (size_t)(1 * DDEC + j) * DDEC;
    const float* wi2 = Wih + (size_t)(2 * DDEC + j) * DDEC;
    const float* wi3 = Wih + (size_t)(3 * DDEC + j) * DDEC;
    const float* wh0 = Whh + (size_t)(0 * DDEC + j) * DDEC;
    const float* wh1 = Whh + (size_t)(1 * DDEC + j) * DDEC;
    const float* wh2 = Whh + (size_t)(2 * DDEC + j) * DDEC;
    const float* wh3 = Whh + (size_t)(3 * DDEC + j) * DDEC;
    for (int t = 0; t < U_; ++t) {
        xt[j] = x[(size_t)(b * U_ + t) * DDEC + j];
        hs[j] = hprev;
        __syncthreads();
        float gi = b0, gf = b1, gg = b2, go = b3;
        for (int k = 0; k < DDEC; ++k) {
            float xv = xt[k], hv = hs[k];
            gi += xv * wi0[k] + hv * wh0[k];
            gf += xv * wi1[k] + hv * wh1[k];
            gg += xv * wi2[k] + hv * wh2[k];
            go += xv * wi3[k] + hv * wh3[k];
        }
        float si = 1.f / (1.f + expf(-gi));
        float sf = 1.f / (1.f + expf(-gf));
        float so = 1.f / (1.f + expf(-go));
        c = sf * c + si * tanhf(gg);
        float h = so * tanhf(c);
        hprev = h;
        out[(size_t)(b * U_ + t) * DDEC + j] = h;
        __syncthreads();
    }
}

// -------------------------------------------------------- encoder projection
__global__ void __launch_bounds__(DJ) k_encp(const float* __restrict__ hs_pad,
        const float* __restrict__ W, const float* __restrict__ bias,
        float* __restrict__ enc_p) {
    __shared__ float row[DENC];
    int r = blockIdx.x;                 // B*T rows
    int j = threadIdx.x;
    row[j]       = hs_pad[(size_t)r * DENC + j];
    row[j + 256] = hs_pad[(size_t)r * DENC + j + 256];
    __syncthreads();
    const float* w = W + (size_t)j * DENC;
    float acc = bias[j];
    for (int k = 0; k < DENC; ++k) acc += row[k] * w[k];
    enc_p[(size_t)r * DJ + j] = acc;
}

// -------------------------------------------------------- decoder projection
__global__ void __launch_bounds__(DJ) k_decp(const float* __restrict__ dec_out,
        const float* __restrict__ W, float* __restrict__ dec_p) {
    __shared__ float row[DDEC];
    int r = blockIdx.x;                 // B*U rows
    int j = threadIdx.x;
    row[j] = dec_out[(size_t)r * DDEC + j];
    __syncthreads();
    const float* w = W + (size_t)j * DDEC;
    float acc = 0.f;
    for (int k = 0; k < DDEC; ++k) acc += row[k] * w[k];
    dec_p[(size_t)r * DJ + j] = acc;
}

// ---------------------------------------------------- weight fp32 -> f16 copy
__global__ void k_cvt(const float* __restrict__ W, _Float16* __restrict__ Wh) {
    int i = blockIdx.x * 256 + threadIdx.x;
    Wh[i] = (_Float16)W[i];
}

// ------------------------------------------------------------- joint (WMMA)
// Block: 256 threads = 8 waves, one 32-row M tile (rows of flattened (b,t,u)).
// Wave w owns N columns [w*64, w*64+64) for BOTH 16-row M subtiles, processed
// as two passes of 2 N-tiles each: every B (weight) fragment is read once per
// wave and feeds two WMMAs, while live accumulators stay at 2x2x8 = 32 VGPRs
// (keeps the kernel under the 256-VGPR boundary -> no s_set_vgpr_msb churn).
// K = 256 in 8 chunks of 32 (v_wmma_f32_16x16x32_f16, fp32 accumulate).
#define MTILE 32
__global__ void __launch_bounds__(256) k_joint(const float* __restrict__ enc_p,
        const float* __restrict__ dec_p, const _Float16* __restrict__ Wh,
        const float* __restrict__ bias, float* __restrict__ jout) {
    const int ASTR = 264;                       // padded half-stride (bank spread)
    __shared__ _Float16 At[MTILE * ASTR];
    int wg  = blockIdx.x;
    int tid = threadIdx.x;

    // Phase 1: 32x256 tanh(enc+dec) tile into LDS (f16); 32 elems/thread
    {
        int m  = tid >> 3;                       // 0..31
        int kb = (tid & 7) << 5;                 // 0..224 step 32
        int gr  = wg * MTILE + m;
        int b   = gr >> 14;                      // T*U = 16384
        int rem = gr & 16383;
        int t   = rem >> 6;
        int u   = rem & 63;
        const float* ep = enc_p + (size_t)(b * T_ + t) * DJ + kb;
        const float* dp = dec_p + (size_t)(b * U_ + u) * DJ + kb;
        _Float16* dst = At + m * ASTR + kb;
        #pragma unroll
        for (int j = 0; j < 32; ++j) dst[j] = (_Float16)tanhf(ep[j] + dp[j]);
    }
    __syncthreads();

    int wv   = tid >> 5;
    int lane = tid & 31;
    int half = lane >> 4;
    int ln   = lane & 15;
    int n0   = wv * 64;

    #pragma unroll
    for (int np = 0; np < 2; ++np) {            // pair of N-tiles per pass
        // Accumulators: 2 M-subtiles x 2 N-tiles, init with bias
        v8f acc[2][2];
        #pragma unroll
        for (int nti = 0; nti < 2; ++nti) {
            float bv = bias[n0 + (np * 2 + nti) * 16 + ln];
            #pragma unroll
            for (int ms = 0; ms < 2; ++ms)
                #pragma unroll
                for (int i = 0; i < 8; ++i) acc[ms][nti][i] = bv;
        }

        #pragma unroll
        for (int kc = 0; kc < 8; ++kc) {
            // A fragments: lanes 0-15 hold K 0..15, lanes 16-31 hold K 16..31
            // within each K-chunk of 32 (matches B).
            v16h afr[2];
            #pragma unroll
            for (int ms = 0; ms < 2; ++ms) {
                const _Float16* arow = At + (ms * 16 + ln) * ASTR + kc * 32 + half * 16;
                v8h a0 = *(const v8h*)(arow);
                v8h a1 = *(const v8h*)(arow + 8);
                v16h a;
                #pragma unroll
                for (int j = 0; j < 8; ++j) { a[j] = a0[j]; a[8 + j] = a1[j]; }
                afr[ms] = a;
            }
            #pragma unroll
            for (int nti = 0; nti < 2; ++nti) {
                // B layout: lane = column n; lanes 0-15 K 0..15, 16-31 K 16..31
                int n = n0 + (np * 2 + nti) * 16 + ln;
                v16h bf = *(const v16h*)(Wh + (size_t)n * DJ + kc * 32 + half * 16);
                acc[0][nti] = __builtin_amdgcn_wmma_f32_16x16x32_f16(
                    false, afr[0], false, bf, (short)0, acc[0][nti], false, false);
                acc[1][nti] = __builtin_amdgcn_wmma_f32_16x16x32_f16(
                    false, afr[1], false, bf, (short)0, acc[1][nti], false, false);
            }
        }

        // C/D layout: VGPR i -> row i (lanes 0-15) / row i+8 (lanes 16-31)
        #pragma unroll
        for (int ms = 0; ms < 2; ++ms) {
            int grbase = wg * MTILE + ms * 16 + half * 8;
            #pragma unroll
            for (int nti = 0; nti < 2; ++nti) {
                int n = n0 + (np * 2 + nti) * 16 + ln;
                #pragma unroll
                for (int i = 0; i < 8; ++i)
                    jout[(size_t)(grbase + i) * V_ + n] = acc[ms][nti][i];
            }
        }
    }
}

// ---------------------------------------------- fused log-sum-exp + extraction
__global__ void __launch_bounds__(256) k_lse(const float* __restrict__ jout,
        const int* __restrict__ ys, float* __restrict__ lpb,
        float* __restrict__ lpl) {
    int gr   = blockIdx.x * 8 + (threadIdx.x >> 5);   // one wave per row
    int lane = threadIdx.x & 31;
    const float* p = jout + (size_t)gr * V_ + lane * 16;
    float v[16];
    #pragma unroll
    for (int j = 0; j < 16; ++j) v[j] = p[j];
    float mx = v[0];
    #pragma unroll
    for (int j = 1; j < 16; ++j) mx = fmaxf(mx, v[j]);
    #pragma unroll
    for (int off = 16; off >= 1; off >>= 1) mx = fmaxf(mx, __shfl_xor(mx, off, 32));
    float s = 0.f;
    #pragma unroll
    for (int j = 0; j < 16; ++j) s += expf(v[j] - mx);
    #pragma unroll
    for (int off = 16; off >= 1; off >>= 1) s += __shfl_xor(s, off, 32);
    float lse = mx + logf(s);
    if (lane == 0) {
        lpb[gr] = jout[(size_t)gr * V_] - lse;              // BLANK = 0
        int u = gr & 63;
        if (u < L_) {
            int b = gr >> 14;
            int tgt = ys[b * L_ + u];
            lpl[gr] = jout[(size_t)gr * V_ + tgt] - lse;
        }
    }
}

// -------------------------------------------------- alpha recursion + reduce
__global__ void k_alpha(const float* __restrict__ lpb, const float* __restrict__ lpl,
                        float* __restrict__ loss_out) {
    __shared__ float ls[B_];
    int b = threadIdx.x;
    if (b < B_) {
        float a[U_];
        a[0] = 0.f;
        for (int u = 1; u < U_; ++u)
            a[u] = a[u - 1] + lpl[(size_t)(b * T_ + 0) * U_ + (u - 1)];
        for (int t = 1; t < T_; ++t) {
            const float* bp = lpb + (size_t)(b * T_ + (t - 1)) * U_;
            const float* lp = lpl + (size_t)(b * T_ + t) * U_;
            float tmp[U_];
            for (int u = 0; u < U_; ++u) tmp[u] = a[u] + bp[u];
            a[0] = tmp[0];
            for (int u = 1; u < U_; ++u) {
                float x = tmp[u];
                float y = a[u - 1] + lp[u - 1];
                float m = fmaxf(x, y);
                a[u] = m + log1pf(expf(-fabsf(x - y)));
            }
        }
        ls[b] = -(a[U_ - 1] + lpb[(size_t)(b * T_ + (T_ - 1)) * U_ + (U_ - 1)]);
    }
    __syncthreads();
    if (threadIdx.x == 0) {
        float s = 0.f;
        for (int i = 0; i < B_; ++i) s += ls[i];
        loss_out[0] = s / (float)(B_ * B_);   // mean(losses)/B
    }
}

extern "C" void kernel_launch(void* const* d_in, const int* in_sizes, int n_in,
                              void* d_out, int out_size, void* d_ws, size_t ws_size,
                              hipStream_t stream) {
    (void)in_sizes; (void)n_in; (void)out_size; (void)ws_size;
    const float* hs_pad    = (const float*)d_in[0];
    const int*   ys_pad    = (const int*)d_in[2];
    const float* embed     = (const float*)d_in[4];
    const float* Wih0      = (const float*)d_in[5];
    const float* Whh0      = (const float*)d_in[6];
    const float* bih0      = (const float*)d_in[7];
    const float* bhh0      = (const float*)d_in[8];
    const float* Wih1      = (const float*)d_in[9];
    const float* Whh1      = (const float*)d_in[10];
    const float* bih1      = (const float*)d_in[11];
    const float* bhh1      = (const float*)d_in[12];
    const float* lin_enc_w = (const float*)d_in[13];
    const float* lin_enc_b = (const float*)d_in[14];
    const float* lin_dec_w = (const float*)d_in[15];
    const float* lin_out_w = (const float*)d_in[16];
    const float* lin_out_b = (const float*)d_in[17];

    float* out  = (float*)d_out;
    float* jout = out + 1;                 // joint_out follows the scalar loss

    float* ws   = (float*)d_ws;
    float* x0   = ws;                                  // B*U*DDEC
    float* h1   = x0   + (size_t)B_ * U_ * DDEC;
    float* h2   = h1   + (size_t)B_ * U_ * DDEC;
    float* encp = h2   + (size_t)B_ * U_ * DDEC;       // B*T*DJ
    float* decp = encp + (size_t)B_ * T_ * DJ;         // B*U*DJ
    float* lpb  = decp + (size_t)B_ * U_ * DJ;         // B*T*U
    float* lpl  = lpb  + (size_t)B_ * T_ * U_;         // B*T*U
    _Float16* Wh = (_Float16*)(lpl + (size_t)B_ * T_ * U_);  // V*DJ halfs

    k_embed<<<(B_ * U_ * DDEC) / 256, 256, 0, stream>>>(embed, ys_pad, x0);
    k_lstm <<<B_, DDEC, 0, stream>>>(x0, Wih0, Whh0, bih0, bhh0, h1);
    k_lstm <<<B_, DDEC, 0, stream>>>(h1, Wih1, Whh1, bih1, bhh1, h2);
    k_encp <<<B_ * T_, DJ, 0, stream>>>(hs_pad, lin_enc_w, lin_enc_b, encp);
    k_decp <<<B_ * U_, DJ, 0, stream>>>(h2, lin_dec_w, decp);
    k_cvt  <<<(V_ * DJ) / 256, 256, 0, stream>>>(lin_out_w, Wh);
    k_joint<<<(B_ * T_ * U_) / MTILE, 256, 0, stream>>>(encp, decp, Wh, lin_out_b, jout);
    k_lse  <<<(B_ * T_ * U_) / 8, 256, 0, stream>>>(jout, ys_pad, lpb, lpl);
    k_alpha<<<1, 32, 0, stream>>>(lpb, lpl, out);
}
